// G2MoEAttention_54863912239456
// MI455X (gfx1250) — compile-verified, matrix operand order checked by reading, not run on hardware
//
#include <hip/hip_runtime.h>
#include <hip/hip_bf16.h>

#define S_LEN   4096
#define HIDDEN  2048
#define NH      16
#define NKV     8
#define DH      128
#define WINDOW  1024
#define SOFTCAP 50.0f
#define SCALING 0.08838834764831845f /* 128^-0.5 */

typedef __attribute__((ext_vector_type(16))) __bf16 bf16x16;
typedef __attribute__((ext_vector_type(8)))  __bf16 bf16x8;
typedef __attribute__((ext_vector_type(8)))  float  floatx8;
typedef __attribute__((ext_vector_type(4)))  unsigned int u32x4;
typedef __attribute__((ext_vector_type(8)))  int i32x8;
typedef __attribute__((ext_vector_type(4)))  int i32x4;

#if defined(__HIP_DEVICE_COMPILE__) && __has_builtin(__builtin_amdgcn_tensor_load_to_lds) && \
    __has_builtin(__builtin_amdgcn_s_wait_tensorcnt)
#define GEMM_TDM 1
#endif

__device__ __forceinline__ floatx8 zero8() {
  floatx8 z;
#pragma unroll
  for (int i = 0; i < 8; ++i) z[i] = 0.0f;
  return z;
}

// Load a 16x32 bf16 A/B fragment from a row-major [16][strideElems] tile.
// CDNA5 16-bit fragment layout: lane l covers row (l&15); regs 0-3 hold K
// chunk ((l>>4)*8 .. +7), regs 4-7 hold K chunk (16 + (l>>4)*8 .. +7).
// Row-major storage => two contiguous 16-byte loads per lane.
__device__ __forceinline__ bf16x16 load_frag(const __bf16* base, int strideElems, int lane) {
  const int row  = lane & 15;
  const int half = lane >> 4;
  const __bf16* p = base + (size_t)row * strideElems + half * 8;
  bf16x8 lo = *(const bf16x8*)(p);
  bf16x8 hi = *(const bf16x8*)(p + 16);
  bf16x16 f;
#pragma unroll
  for (int i = 0; i < 8; ++i) { f[i] = lo[i]; f[i + 8] = hi[i]; }
  return f;
}

// ---------------------------------------------------------------- casts
__global__ void cast_f32_to_bf16(const float* __restrict__ src, __bf16* __restrict__ dst, int n) {
  int i = blockIdx.x * blockDim.x + threadIdx.x;
  int stride = gridDim.x * blockDim.x;
  for (; i < n; i += stride) dst[i] = (__bf16)src[i];
}

// ---------------------------------------------------------------- GEMM
// C[M,N] = A[M,K] @ B[K,N], bf16 in, OutT out. Block tile 128x64, 8 waves,
// each wave computes 2x2 WMMA 16x16 tiles, K-step 32 staged through LDS.
// A tile staged by the Tensor Data Mover (TDM) when available; B tile is
// transposed into LDS by the threads (TDM cannot transpose).
template <typename OutT>
__global__ __launch_bounds__(256) void gemm_bf16(const __bf16* __restrict__ A,
                                                 const __bf16* __restrict__ B,
                                                 OutT* __restrict__ C,
                                                 int M, int N, int K) {
  __shared__ __align__(16) __bf16 As[128 * 32];  // [row][k]
  __shared__ __align__(16) __bf16 Bs[64 * 32];   // transposed: [n][k]
  const int tid  = threadIdx.x;
  const int lane = tid & 31;
  const int wave = tid >> 5;
  const int wm = wave >> 1;  // 0..3
  const int wn = wave & 1;   // 0..1
  const int m0 = blockIdx.x * 128;
  const int n0 = blockIdx.y * 64;

  floatx8 acc[2][2];
#pragma unroll
  for (int i = 0; i < 2; ++i)
#pragma unroll
    for (int j = 0; j < 2; ++j) acc[i][j] = zero8();

  for (int k0 = 0; k0 < K; k0 += 32) {
#ifdef GEMM_TDM
    // --- Tensor Data Mover staging of the 128x32 A tile (2D descriptor) ---
    if (wave == 0) {
      unsigned long long ga = (unsigned long long)(uintptr_t)(A + (size_t)m0 * K + k0);
      // Generic LDS pointer: low 32 bits are the wave-relative LDS offset
      // (ISA 10.2).
      unsigned lds_base = (unsigned)(uintptr_t)(const void*)&As[0];
      u32x4 g0;
      g0[0] = 1u;                                 // count=1 (valid user D#)
      g0[1] = lds_base;                           // lds_addr
      g0[2] = (unsigned)ga;                       // global_addr[31:0]
      g0[3] = (unsigned)(ga >> 32) | (2u << 30);  // global_addr[56:32] | type=2
      i32x8 g1;
      g1[0] = 1 << 16;                            // data_size=1 (2B/elem)
      g1[1] = (int)(((unsigned)K & 0xFFFFu) << 16);                      // tensor_dim0 lo16
      g1[2] = (int)(((unsigned)K >> 16) | (((unsigned)M & 0xFFFFu) << 16)); // dim0 hi | dim1 lo
      g1[3] = (int)(((unsigned)M >> 16) | (32u << 16));                  // dim1 hi | tile_dim0=32
      g1[4] = 128;                                // tile_dim1=128, tile_dim2=0
      g1[5] = K;                                  // tensor_dim0_stride[31:0]
      g1[6] = 0;                                  // stride0 hi | dim1_stride lo
      g1[7] = 0;
      i32x4 gz = {0, 0, 0, 0};
#if __clang_major__ >= 23
      i32x8 gz8 = {0, 0, 0, 0, 0, 0, 0, 0};
      __builtin_amdgcn_tensor_load_to_lds(g0, g1, gz, gz, gz8, 0);
#else
      __builtin_amdgcn_tensor_load_to_lds(g0, g1, gz, gz, 0);
#endif
    }
#else
    // --- fallback: manual staging of A: 128x32 (2 threads/row, 16 bf16 ea) ---
    {
      int row = tid >> 1;
      int ch  = (tid & 1) * 16;
      const __bf16* src = A + (size_t)(m0 + row) * K + k0 + ch;
      if (k0 + 32 < K) __builtin_prefetch(src + 32, 0, 1);
      bf16x8 v0 = *(const bf16x8*)(src);
      bf16x8 v1 = *(const bf16x8*)(src + 8);
      *(bf16x8*)&As[row * 32 + ch]     = v0;
      *(bf16x8*)&As[row * 32 + ch + 8] = v1;
    }
#endif
    // stage B transposed: Bs[n][k] = B[k0+k][n0+n] (overlaps with TDM)
    {
      int k  = tid >> 3;        // 0..31
      int ng = (tid & 7) * 8;   // 0..56
      const __bf16* src = B + (size_t)(k0 + k) * N + n0 + ng;
      if (k0 + 32 < K) __builtin_prefetch(src + (size_t)32 * N, 0, 1);
      bf16x8 v = *(const bf16x8*)src;
#pragma unroll
      for (int i = 0; i < 8; ++i) Bs[(ng + i) * 32 + k] = v[i];
    }
#ifdef GEMM_TDM
    if (wave == 0) __builtin_amdgcn_s_wait_tensorcnt(0);
#endif
    __syncthreads();

    bf16x16 af[2], bfr[2];
#pragma unroll
    for (int i = 0; i < 2; ++i) af[i]  = load_frag(&As[(wm * 32 + i * 16) * 32], 32, lane);
#pragma unroll
    for (int j = 0; j < 2; ++j) bfr[j] = load_frag(&Bs[(wn * 32 + j * 16) * 32], 32, lane);
#pragma unroll
    for (int i = 0; i < 2; ++i)
#pragma unroll
      for (int j = 0; j < 2; ++j)
        acc[i][j] = __builtin_amdgcn_wmma_f32_16x16x32_bf16(
            false, af[i], false, bfr[j], (short)0, acc[i][j], false, false);
    __syncthreads();
  }

  // C/D layout: reg r, lanes 0-15 -> M=r, N=lane; lanes 16-31 -> M=8+r.
  const int cn  = lane & 15;
  const int rhi = (lane >> 4) * 8;
#pragma unroll
  for (int i = 0; i < 2; ++i)
#pragma unroll
    for (int j = 0; j < 2; ++j)
#pragma unroll
      for (int r = 0; r < 8; ++r) {
        int mrow = m0 + wm * 32 + i * 16 + rhi + r;
        int ncol = n0 + wn * 32 + j * 16 + cn;
        C[(size_t)mrow * N + ncol] = (OutT)acc[i][j][r];
      }
}

// ---------------------------------------------------------------- RoPE
// in: [S][nh*128] bf16 ; out: [nh][S][128] bf16. cos[s][d]==cos[s][d+64].
__global__ void rope_kernel(const __bf16* __restrict__ in, const float* __restrict__ cosp,
                            const float* __restrict__ sinp, __bf16* __restrict__ outp, int nh) {
  int idx = blockIdx.x * blockDim.x + threadIdx.x;
  int total = nh * S_LEN * (DH / 2);
  if (idx >= total) return;
  int d = idx % (DH / 2);
  int s = (idx / (DH / 2)) % S_LEN;
  int h = idx / ((DH / 2) * S_LEN);
  size_t rowoff = (size_t)s * (nh * DH) + h * DH;
  float x1 = (float)in[rowoff + d];
  float x2 = (float)in[rowoff + d + 64];
  float c  = cosp[s * DH + d];
  float sn = sinp[s * DH + d];
  size_t o = ((size_t)h * S_LEN + s) * DH;
  outp[o + d]      = (__bf16)(x1 * c - x2 * sn);
  outp[o + d + 64] = (__bf16)(x2 * c + x1 * sn);
}

// V: [S][NKV*128] -> Vt: [NKV][128][S]
__global__ void vtrans_kernel(const __bf16* __restrict__ Vp, __bf16* __restrict__ Vt) {
  int idx = blockIdx.x * blockDim.x + threadIdx.x;
  int total = NKV * DH * S_LEN;
  if (idx >= total) return;
  int s  = idx % S_LEN;
  int d  = (idx / S_LEN) % DH;
  int kv = idx / (S_LEN * DH);
  Vt[idx] = Vp[(size_t)s * (NKV * DH) + kv * DH + d];
}

// ---------------------------------------------------------------- attention
// One wave per (head, 16-query tile). 32 keys/iter: 8 WMMA for QK^T,
// 8 WMMA for PV. Sliding window + causal mask computed from indices.
__global__ __launch_bounds__(256) void attn_kernel(const __bf16* __restrict__ Qr,
                                                   const __bf16* __restrict__ Kr,
                                                   const __bf16* __restrict__ Vt,
                                                   __bf16* __restrict__ AO) {
  __shared__ __align__(16) __bf16 Plds[8][16 * 32];
  const int lane = threadIdx.x & 31;
  const int wv   = threadIdx.x >> 5;
  const int wid  = blockIdx.x * 8 + wv;
  const int h    = wid >> 8;           // wid / (S/16)
  const int q0   = (wid & 255) << 4;
  const int kvh  = h >> 1;             // N_REP = 2
  const int n    = lane & 15;
  const int hi8  = (lane >> 4) << 3;

  const __bf16* Qbase = Qr + ((size_t)h * S_LEN + q0) * DH;
  const __bf16* Kbase = Kr + (size_t)kvh * S_LEN * DH;
  const __bf16* Vbase = Vt + (size_t)kvh * DH * S_LEN;

  bf16x16 qf[4];
#pragma unroll
  for (int c = 0; c < 4; ++c) qf[c] = load_frag(Qbase + c * 32, DH, lane);

  floatx8 acc[8];
#pragma unroll
  for (int dc = 0; dc < 8; ++dc) acc[dc] = zero8();
  float mrow[8], lrow[8];
#pragma unroll
  for (int r = 0; r < 8; ++r) { mrow[r] = -3.0e38f; lrow[r] = 0.0f; }

  const float NEG_INF = -__builtin_inff();
  int kstart = (q0 >= WINDOW) ? ((q0 - (WINDOW - 1)) & ~31) : 0;

  for (int k0 = kstart; k0 <= q0 + 15; k0 += 32) {
    floatx8 s0 = zero8(), s1 = zero8();
#pragma unroll
    for (int c = 0; c < 4; ++c) {
      bf16x16 kf0 = load_frag(Kbase + (size_t)k0 * DH + c * 32, DH, lane);
      bf16x16 kf1 = load_frag(Kbase + (size_t)(k0 + 16) * DH + c * 32, DH, lane);
      s0 = __builtin_amdgcn_wmma_f32_16x16x32_bf16(false, qf[c], false, kf0, (short)0, s0, false, false);
      s1 = __builtin_amdgcn_wmma_f32_16x16x32_bf16(false, qf[c], false, kf1, (short)0, s1, false, false);
    }
    float corr[8], e0[8], e1[8];
#pragma unroll
    for (int r = 0; r < 8; ++r) {
      int i  = q0 + hi8 + r;
      int j0 = k0 + n;
      int j1 = k0 + 16 + n;
      float x0 = SOFTCAP * tanhf(s0[r] * (SCALING / SOFTCAP));
      float x1 = SOFTCAP * tanhf(s1[r] * (SCALING / SOFTCAP));
      if (j0 > i || (i - j0) >= WINDOW) x0 = NEG_INF;
      if (j1 > i || (i - j1) >= WINDOW) x1 = NEG_INF;
      float t = fmaxf(x0, x1);
#pragma unroll
      for (int msk = 1; msk < 16; msk <<= 1) t = fmaxf(t, __shfl_xor(t, msk, 32));
      float mnew = fmaxf(mrow[r], t);
      float cr = __expf(mrow[r] - mnew);
      float a0 = __expf(x0 - mnew);
      float a1 = __expf(x1 - mnew);
      float ls = a0 + a1;
#pragma unroll
      for (int msk = 1; msk < 16; msk <<= 1) ls += __shfl_xor(ls, msk, 32);
      lrow[r] = lrow[r] * cr + ls;
      mrow[r] = mnew;
      corr[r] = cr; e0[r] = a0; e1[r] = a1;
    }
#pragma unroll
    for (int dc = 0; dc < 8; ++dc)
#pragma unroll
      for (int r = 0; r < 8; ++r) acc[dc][r] *= corr[r];

    // stage P (C-layout -> row-major 16x32) through this wave's LDS slice
#pragma unroll
    for (int r = 0; r < 8; ++r) {
      Plds[wv][(hi8 + r) * 32 + n]      = (__bf16)e0[r];
      Plds[wv][(hi8 + r) * 32 + 16 + n] = (__bf16)e1[r];
    }
    asm volatile("s_wait_dscnt 0" ::: "memory");  // within-wave RAW on LDS
    bf16x16 pf = load_frag(&Plds[wv][0], 32, lane);
#pragma unroll
    for (int dc = 0; dc < 8; ++dc) {
      bf16x16 vf = load_frag(Vbase + (size_t)(dc * 16) * S_LEN + k0, S_LEN, lane);
      acc[dc] = __builtin_amdgcn_wmma_f32_16x16x32_bf16(false, pf, false, vf, (short)0, acc[dc], false, false);
    }
  }

#pragma unroll
  for (int dc = 0; dc < 8; ++dc)
#pragma unroll
    for (int r = 0; r < 8; ++r) {
      int row = q0 + hi8 + r;
      float v = acc[dc][r] / lrow[r];
      AO[(size_t)row * (NH * DH) + h * DH + dc * 16 + n] = (__bf16)v;
    }
}

// ---------------------------------------------------------------- launch
extern "C" void kernel_launch(void* const* d_in, const int* in_sizes, int n_in,
                              void* d_out, int out_size, void* d_ws, size_t ws_size,
                              hipStream_t stream) {
  (void)in_sizes; (void)n_in; (void)out_size; (void)ws_size;
  const float* hs   = (const float*)d_in[0];
  const float* cosp = (const float*)d_in[1];
  const float* sinp = (const float*)d_in[2];
  /* d_in[3] attention_mask: recomputed analytically, never read */
  const float* Wq   = (const float*)d_in[4];
  const float* Wk   = (const float*)d_in[5];
  const float* Wv   = (const float*)d_in[6];
  const float* Wo   = (const float*)d_in[7];
  float* out = (float*)d_out;

  char* ws = (char*)d_ws;
  size_t off = 0;
  auto wsalloc = [&](size_t bytes) -> char* {
    char* p = ws + off;
    off += (bytes + 255) & ~(size_t)255;
    return p;
  };
  __bf16* hsb = (__bf16*)wsalloc((size_t)S_LEN * HIDDEN * 2);
  __bf16* Wqb = (__bf16*)wsalloc((size_t)HIDDEN * NH * DH * 2);
  __bf16* Wkb = (__bf16*)wsalloc((size_t)HIDDEN * NKV * DH * 2);
  __bf16* Wvb = (__bf16*)wsalloc((size_t)HIDDEN * NKV * DH * 2);
  __bf16* Wob = (__bf16*)wsalloc((size_t)NH * DH * HIDDEN * 2);
  __bf16* Qp  = (__bf16*)wsalloc((size_t)S_LEN * NH * DH * 2);
  __bf16* Kp  = (__bf16*)wsalloc((size_t)S_LEN * NKV * DH * 2);
  __bf16* Vp  = (__bf16*)wsalloc((size_t)S_LEN * NKV * DH * 2);
  __bf16* Qrb = (__bf16*)wsalloc((size_t)NH * S_LEN * DH * 2);
  __bf16* Krb = (__bf16*)wsalloc((size_t)NKV * S_LEN * DH * 2);
  __bf16* Vtb = (__bf16*)wsalloc((size_t)NKV * DH * S_LEN * 2);
  __bf16* AOb = (__bf16*)wsalloc((size_t)S_LEN * NH * DH * 2);

  const int T = 256;
  cast_f32_to_bf16<<<4096, T, 0, stream>>>(hs, hsb, S_LEN * HIDDEN);
  cast_f32_to_bf16<<<4096, T, 0, stream>>>(Wq, Wqb, HIDDEN * NH * DH);
  cast_f32_to_bf16<<<2048, T, 0, stream>>>(Wk, Wkb, HIDDEN * NKV * DH);
  cast_f32_to_bf16<<<2048, T, 0, stream>>>(Wv, Wvb, HIDDEN * NKV * DH);
  cast_f32_to_bf16<<<4096, T, 0, stream>>>(Wo, Wob, NH * DH * HIDDEN);

  gemm_bf16<__bf16><<<dim3(S_LEN / 128, (NH * DH) / 64), T, 0, stream>>>(
      hsb, Wqb, Qp, S_LEN, NH * DH, HIDDEN);
  gemm_bf16<__bf16><<<dim3(S_LEN / 128, (NKV * DH) / 64), T, 0, stream>>>(
      hsb, Wkb, Kp, S_LEN, NKV * DH, HIDDEN);
  gemm_bf16<__bf16><<<dim3(S_LEN / 128, (NKV * DH) / 64), T, 0, stream>>>(
      hsb, Wvb, Vp, S_LEN, NKV * DH, HIDDEN);

  rope_kernel<<<(NH * S_LEN * (DH / 2)) / T, T, 0, stream>>>(Qp, cosp, sinp, Qrb, NH);
  rope_kernel<<<(NKV * S_LEN * (DH / 2)) / T, T, 0, stream>>>(Kp, cosp, sinp, Krb, NKV);
  vtrans_kernel<<<(NKV * DH * S_LEN) / T, T, 0, stream>>>(Vp, Vtb);

  attn_kernel<<<(NH * (S_LEN / 16)) / 8, T, 0, stream>>>(Qrb, Krb, Vtb, AOb);

  gemm_bf16<float><<<dim3(S_LEN / 128, HIDDEN / 64), T, 0, stream>>>(
      AOb, Wob, out, S_LEN, HIDDEN, NH * DH);
}
